// MeshFC_43748536877517
// MI455X (gfx1250) — matrix operand order repaired and checked by reading, not compiled, not run on hardware
//
#include <hip/hip_runtime.h>
#include <hip/hip_bf16.h>

// ---- problem constants (from the reference) ----
#define NUM_IN   2048   // I (GEMM K)
#define NUM_OUT  8192   // O (GEMM N)
#define BATCH    4096   // B (GEMM M)
#define SD       5

typedef _Float16 half8  __attribute__((ext_vector_type(8)));
typedef _Float16 half16 __attribute__((ext_vector_type(16)));
typedef float    f32x4  __attribute__((ext_vector_type(4)));
typedef float    f32x8  __attribute__((ext_vector_type(8)));

// -------------------------------------------------------------------------
// Kernel 1: convert inputs f32 -> f16 (Af[B, I]), 8 elements per thread.
// Source data is read exactly once -> non-temporal loads.
// -------------------------------------------------------------------------
__global__ __launch_bounds__(256) void meshfc_cvt_f16(
    const float* __restrict__ in, _Float16* __restrict__ out) {
  size_t idx = ((size_t)blockIdx.x * blockDim.x + threadIdx.x) * 8;
  const f32x4* p = (const f32x4*)(in + idx);
  f32x4 a = __builtin_nontemporal_load(p);
  f32x4 b = __builtin_nontemporal_load(p + 1);
  half8 h;
  h[0] = (_Float16)a[0]; h[1] = (_Float16)a[1];
  h[2] = (_Float16)a[2]; h[3] = (_Float16)a[3];
  h[4] = (_Float16)b[0]; h[5] = (_Float16)b[1];
  h[6] = (_Float16)b[2]; h[7] = (_Float16)b[3];
  *(half8*)(out + idx) = h;   // Af is re-read by the GEMM: keep RT (cache it)
}

// -------------------------------------------------------------------------
// Kernel 2: build transposed weight matrix Wt[O, I] in f16.
//   w[i,o] = ||in_pos[i]-out_pos[o]|| - ||init_in[i]-init_out[o]||
// One block per output o; each thread handles 8 consecutive i.
// -------------------------------------------------------------------------
__global__ __launch_bounds__(256) void meshfc_weights(
    const float* __restrict__ in_pos,       // [I,5]
    const float* __restrict__ out_pos,      // [O,5]
    const float* __restrict__ init_in_pos,  // [I,5]
    const float* __restrict__ init_out_pos, // [O,5]
    _Float16* __restrict__ Wt) {            // [O, I]
  int o  = blockIdx.x;
  int i0 = threadIdx.x * 8;

  float op[SD], iop[SD];
#pragma unroll
  for (int d = 0; d < SD; ++d) {
    op[d]  = out_pos[(size_t)o * SD + d];
    iop[d] = init_out_pos[(size_t)o * SD + d];
  }

  half8 res;
#pragma unroll
  for (int k = 0; k < 8; ++k) {
    int i = i0 + k;
    float d1 = 0.f, d0 = 0.f;
#pragma unroll
    for (int d = 0; d < SD; ++d) {
      float t1 = in_pos[(size_t)i * SD + d] - op[d];
      float t0 = init_in_pos[(size_t)i * SD + d] - iop[d];
      d1 = fmaf(t1, t1, d1);
      d0 = fmaf(t0, t0, d0);
    }
    res[k] = (_Float16)(__builtin_sqrtf(d1) - __builtin_sqrtf(d0));
  }
  *(half8*)(Wt + (size_t)o * NUM_IN + i0) = res;  // re-read by GEMM: keep RT
}

// -------------------------------------------------------------------------
// Kernel 3: GEMM  out[B,O] = Af[B,I] * Wt[O,I]^T + bias
// Block tile 128(M) x 256(N); 8 waves (2x4); wave tile 64x64 = 4x4 WMMA tiles
// -> 16 global_load_b128 : 16 v_wmma per K=32 step (1.0 loads/WMMA).
// All 8 fragments (4 A + 4 B) are preloaded into distinct registers before
// the WMMA block so the in-order load queue gives staggered partial waits
// (loadcnt 3/2/1/0) instead of full drains, and next-iteration loads can be
// hoisted between WMMA groups.
// A fragment (16x32 f16): lane(h=lane>>4, m=lane&15) holds K=16h..16h+15
//   of row m -> two 16B contiguous loads from row-major Af.
// B fragment (32x16 f16): lane holds column n=lane&15, K=16h..16h+15 ->
//   one 32B contiguous load from row n of Wt[O,I].
// C/D (16x16 f32): VGPR j -> row 8h+j, col lane&15.
// -------------------------------------------------------------------------
__global__ __launch_bounds__(256) void meshfc_gemm(
    const _Float16* __restrict__ Af,   // [B, I]
    const _Float16* __restrict__ Wt,   // [O, I]
    const float*    __restrict__ bias, // [O]
    float*          __restrict__ out) {// [B, O]
  const int wave = threadIdx.x >> 5;
  const int lane = threadIdx.x & 31;
  const int h    = lane >> 4;   // which half of the wave
  const int l    = lane & 15;   // row (A) / col (B,C)

  const int wm = wave >> 2;     // 0..1
  const int wn = wave & 3;      // 0..3
  const int rowBase = blockIdx.y * 128 + wm * 64;
  const int colBase = blockIdx.x * 256 + wn * 64;

  f32x8 acc[4][4];
#pragma unroll
  for (int mt = 0; mt < 4; ++mt)
#pragma unroll
    for (int nt = 0; nt < 4; ++nt)
      acc[mt][nt] = (f32x8){0.f, 0.f, 0.f, 0.f, 0.f, 0.f, 0.f, 0.f};

  const _Float16* aBase[4];
#pragma unroll
  for (int mt = 0; mt < 4; ++mt)
    aBase[mt] = Af + (size_t)(rowBase + mt * 16 + l) * NUM_IN + h * 16;
  const _Float16* bBase[4];
#pragma unroll
  for (int nt = 0; nt < 4; ++nt)
    bBase[nt] = Wt + (size_t)(colBase + nt * 16 + l) * NUM_IN + h * 16;

  for (int k0 = 0; k0 < NUM_IN; k0 += 32) {
    // ---- preload all 8 fragments into distinct registers ----
    half16 bfrag[4];
#pragma unroll
    for (int nt = 0; nt < 4; ++nt)
      bfrag[nt] = *(const half16*)(bBase[nt] + k0);   // 32B contiguous

    half16 afrag[4];
#pragma unroll
    for (int mt = 0; mt < 4; ++mt) {
      half8 lo = *(const half8*)(aBase[mt] + k0);       // K = k0+16h .. +7
      half8 hi = *(const half8*)(aBase[mt] + k0 + 8);   // K = k0+16h+8 .. +7
      afrag[mt] = __builtin_shufflevector(
          lo, hi, 0, 1, 2, 3, 4, 5, 6, 7, 8, 9, 10, 11, 12, 13, 14, 15);
    }

    // ---- 16 WMMAs ----
#pragma unroll
    for (int mt = 0; mt < 4; ++mt) {
#pragma unroll
      for (int nt = 0; nt < 4; ++nt) {
        acc[mt][nt] = __builtin_amdgcn_wmma_f32_16x16x32_f16(
            /*neg_a=*/false, afrag[mt], /*neg_b=*/false, bfrag[nt],
            /*c_mod=*/(short)0, acc[mt][nt],
            /*reuse_a=*/false, /*reuse_b=*/false);
      }
    }
  }

  // Epilogue: add bias, store (write-once 128MB -> non-temporal, keep L2
  // for the Af/Wt working set).
  float bv[4];
#pragma unroll
  for (int nt = 0; nt < 4; ++nt)
    bv[nt] = bias[colBase + nt * 16 + l];

#pragma unroll
  for (int mt = 0; mt < 4; ++mt) {
#pragma unroll
    for (int nt = 0; nt < 4; ++nt) {
      const int c = colBase + nt * 16 + l;
#pragma unroll
      for (int j = 0; j < 8; ++j) {
        const int r = rowBase + mt * 16 + h * 8 + j;
        __builtin_nontemporal_store(acc[mt][nt][j] + bv[nt],
                                    out + (size_t)r * NUM_OUT + c);
      }
    }
  }
}

// -------------------------------------------------------------------------
// Host launcher.  setup_inputs order:
//   0: inputs [B,I] f32   1: init_in_pos [I,1,5]   2: init_out_pos [1,O,5]
//   3: in_pos [I,1,5]     4: out_pos [1,O,5]       5: biases [O]
// Workspace: [0, 32MB) Wt f16 [O,I]; [32MB, 48MB) Af f16 [B,I].
// -------------------------------------------------------------------------
extern "C" void kernel_launch(void* const* d_in, const int* in_sizes, int n_in,
                              void* d_out, int out_size, void* d_ws, size_t ws_size,
                              hipStream_t stream) {
  const float* inputs       = (const float*)d_in[0];
  const float* init_in_pos  = (const float*)d_in[1];
  const float* init_out_pos = (const float*)d_in[2];
  const float* in_pos       = (const float*)d_in[3];
  const float* out_pos      = (const float*)d_in[4];
  const float* biases       = (const float*)d_in[5];
  float* out = (float*)d_out;

  _Float16* Wt = (_Float16*)d_ws;
  _Float16* Af = (_Float16*)((char*)d_ws + (size_t)NUM_OUT * NUM_IN * sizeof(_Float16));

  // 1) inputs f32 -> f16 (8 elems/thread)
  {
    const size_t n = (size_t)BATCH * NUM_IN;           // 8,388,608
    const int threads = 256;
    const int blocks  = (int)(n / (threads * 8));      // 4096
    meshfc_cvt_f16<<<blocks, threads, 0, stream>>>(inputs, Af);
  }

  // 2) build Wt[O,I] f16
  meshfc_weights<<<NUM_OUT, 256, 0, stream>>>(in_pos, out_pos,
                                              init_in_pos, init_out_pos, Wt);

  // 3) WMMA GEMM: grid = (N/256, M/128), 256 threads (8 waves, 2x4)
  {
    dim3 grid(NUM_OUT / 256, BATCH / 128);
    meshfc_gemm<<<grid, 256, 0, stream>>>(Af, Wt, biases, out);
  }
}
